// Encoder_15891378995294
// MI455X (gfx1250) — compile-verified, matrix operand order checked by reading, not run on hardware
//
#include <hip/hip_runtime.h>

typedef __attribute__((ext_vector_type(16))) _Float16 v16h;
typedef __attribute__((ext_vector_type(8)))  float    v8f;
typedef __attribute__((ext_vector_type(4)))  float    v4f;

#define B_     16
#define T_     1024
#define C_     1024
#define H_     32
#define W_     32
#define ROWS   (B_*T_)     // 16384
#define NT_KVR 2           // 16x32 tile/wave in fused kvr GEMM (3 accumulators)
#define NT_OUT 4           // 16x64 tile/wave in output GEMM (1 accumulator)

// ---------------- weight f32 -> f16 convert ----------------
__global__ void wcvt_kernel(const float* __restrict__ w, _Float16* __restrict__ o, int n) {
    int i = blockIdx.x * 256 + threadIdx.x;
    if (i < n) o[i] = (_Float16)w[i];
}

// ---------------- fused q_shift + mix + triple WMMA GEMM ----------------
// k = xk @ Wk^T ; v = xv @ Wv^T ; sr = sigmoid(xr @ Wr^T)
// K-loop split per quadrant group so the shift-neighbor selection is hoisted
// out of the hot loop (grp is uniform: c0 multiple of 32 => c0 and c0+16 share
// one 256-channel quadrant). Inner loop is branch-free, EXEC stays full.
__global__ __launch_bounds__(32)
void gemm_kvr_kernel(const float* __restrict__ x,
                     const _Float16* __restrict__ Wk,
                     const _Float16* __restrict__ Wv,
                     const _Float16* __restrict__ Wr,
                     const float* __restrict__ mixk,
                     const float* __restrict__ mixv,
                     const float* __restrict__ mixr,
                     float* __restrict__ kout,
                     float* __restrict__ vout,
                     _Float16* __restrict__ srout)
{
    const int lane    = threadIdx.x;                 // 0..31, wave32
    const int rowBase = blockIdx.x * 16;             // M tile
    const int dBase   = blockIdx.y * (16 * NT_KVR);  // N tile

    const int m     = lane & 15;
    const int khalf = (lane >> 4) << 4;              // 0 or 16 (K sub-chunk)
    const int row   = rowBase + m;
    const int b     = row >> 10;                     // row / T
    const int t     = row & (T_ - 1);
    const int h     = t >> 5;
    const int w     = t & (W_ - 1);

    v8f acc[3][NT_KVR];
    #pragma unroll
    for (int o = 0; o < 3; ++o)
        #pragma unroll
        for (int n = 0; n < NT_KVR; ++n) acc[o][n] = (v8f)0.0f;

    const float* prow = x + (size_t)row * C_;

    #pragma unroll
    for (int grp = 0; grp < 4; ++grp) {
        // quadrant shift target, hoisted out of the K loop (folds at unroll)
        int t2, valid;
        if      (grp == 0) { valid = (w > 0);       t2 = t - 1;  }
        else if (grp == 1) { valid = (w < W_ - 1);  t2 = t + 1;  }
        else if (grp == 2) { valid = (h > 0);       t2 = t - W_; }
        else               { valid = (h < H_ - 1);  t2 = t + W_; }
        const int t2c = t2 < 0 ? 0 : (t2 > T_ - 1 ? T_ - 1 : t2);  // OOB-safe
        const float vf = valid ? 1.0f : 0.0f;
        const float* pshift = x + ((size_t)(b * T_ + t2c)) * C_;

        for (int c8 = 0; c8 < 256; c8 += 32) {
            const int c0 = grp * 256 + c8;
            const int kb = c0 + khalf;               // channel base for this lane

            const float* px  = prow   + kb;
            const float* pxx = pshift + kb;

            v16h ak, av, ar;
            #pragma unroll
            for (int q = 0; q < 4; ++q) {
                v4f xv4  = *(const v4f*)(px  + 4 * q);
                v4f xxv4 = *(const v4f*)(pxx + 4 * q);
                v4f mk4  = *(const v4f*)(mixk + kb + 4 * q);
                v4f mv4  = *(const v4f*)(mixv + kb + 4 * q);
                v4f mr4  = *(const v4f*)(mixr + kb + 4 * q);
                #pragma unroll
                for (int j = 0; j < 4; ++j) {
                    float xs  = xv4[j];
                    float xxs = xxv4[j] * vf;
                    ak[4 * q + j] = (_Float16)(xs * mk4[j] + xxs * (1.0f - mk4[j]));
                    av[4 * q + j] = (_Float16)(xs * mv4[j] + xxs * (1.0f - mv4[j]));
                    ar[4 * q + j] = (_Float16)(xs * mr4[j] + xxs * (1.0f - mr4[j]));
                }
            }

            #pragma unroll
            for (int n = 0; n < NT_KVR; ++n) {
                const int d = dBase + n * 16 + (lane & 15);
                const size_t wo = (size_t)d * C_ + kb;
                v16h bk = *(const v16h*)(Wk + wo);
                v16h bv = *(const v16h*)(Wv + wo);
                v16h br = *(const v16h*)(Wr + wo);
                if (c0 + 32 < C_) {        // warm next K-tile of streamed weights
                    __builtin_prefetch(Wk + wo + 32, 0, 3);
                    __builtin_prefetch(Wv + wo + 32, 0, 3);
                    __builtin_prefetch(Wr + wo + 32, 0, 3);
                }
                acc[0][n] = __builtin_amdgcn_wmma_f32_16x16x32_f16(false, ak, false, bk,
                              (short)0, acc[0][n], false, false);
                acc[1][n] = __builtin_amdgcn_wmma_f32_16x16x32_f16(false, av, false, bv,
                              (short)0, acc[1][n], false, false);
                acc[2][n] = __builtin_amdgcn_wmma_f32_16x16x32_f16(false, ar, false, br,
                              (short)0, acc[2][n], false, false);
            }
        }
    }

    // C/D layout: VGPR i -> M = (lane>>4)*8 + i, N = lane&15
    const int mrow = rowBase + ((lane >> 4) << 3);
    const int ncol = lane & 15;
    #pragma unroll
    for (int n = 0; n < NT_KVR; ++n) {
        #pragma unroll
        for (int i = 0; i < 8; ++i) {
            size_t idx = (size_t)(mrow + i) * C_ + dBase + n * 16 + ncol;
            kout[idx] = acc[0][n][i];
            vout[idx] = acc[1][n][i];
            float rr  = acc[2][n][i];
            srout[idx] = (_Float16)(1.0f / (1.0f + __expf(-rr)));
        }
    }
}

// ---------------- WKV sequential scan over T (per b,c lane) ----------------
// Latency-bound (only B*C/32 = 512 waves): software-pipeline the k/v loads so
// the HBM latency overlaps the dependent exp-chain instead of serializing.
__global__ __launch_bounds__(256)
void wkv_scan_kernel(const float* kin, const float* __restrict__ vin,
                     const float* __restrict__ sd, const float* __restrict__ sf,
                     float* yout /* aliases kin */)
{
    int idx = blockIdx.x * 256 + threadIdx.x;        // 0 .. B*C-1
    int c = idx & (C_ - 1);
    int b = idx >> 10;
    float w_eff = -__expf(sd[c] * (1.0f / T_));
    float u_eff = sf[c] * (1.0f / T_);
    float p = 0.0f, q = 0.0f, o = -1e38f;
    size_t off = (size_t)b * T_ * C_ + c;

    float kt = kin[off];
    float vt = vin[off];
    for (int t = 0; t < T_; ++t) {
        float ktc = kt, vtc = vt;
        if (t + 1 < T_) {                            // preload next step
            kt = kin[off + C_];
            vt = vin[off + C_];
        }
        float no = fmaxf(o, u_eff + ktc);
        float A  = __expf(o - no);
        float Bc = __expf(u_eff + ktc - no);
        float y  = (A * p + Bc * vtc) / (A * q + Bc);
        float no2 = fmaxf(w_eff + o, ktc);
        float A2  = __expf(w_eff + o - no2);
        float B2  = __expf(ktc - no2);
        p = A2 * p + B2 * vtc;
        q = A2 * q + B2;
        o = no2;
        yout[off] = y;
        off += C_;
    }
}

// ---------------- LayerNorm over C + sigmoid gate, f16 output ----------------
__global__ __launch_bounds__(256)
void ln_gate_kernel(const float* __restrict__ y,
                    const _Float16* __restrict__ sr,
                    const float* __restrict__ lnw,
                    const float* __restrict__ lnb,
                    _Float16* __restrict__ z)
{
    __shared__ float red[256];
    const int row = blockIdx.x;
    const float* yr = y + (size_t)row * C_;

    float s = 0.0f;
    for (int c = threadIdx.x; c < C_; c += 256) s += yr[c];
    red[threadIdx.x] = s; __syncthreads();
    for (int st = 128; st > 0; st >>= 1) {
        if (threadIdx.x < st) red[threadIdx.x] += red[threadIdx.x + st];
        __syncthreads();
    }
    float mu = red[0] * (1.0f / C_);
    __syncthreads();

    float s2 = 0.0f;
    for (int c = threadIdx.x; c < C_; c += 256) { float d = yr[c] - mu; s2 += d * d; }
    red[threadIdx.x] = s2; __syncthreads();
    for (int st = 128; st > 0; st >>= 1) {
        if (threadIdx.x < st) red[threadIdx.x] += red[threadIdx.x + st];
        __syncthreads();
    }
    float rstd = rsqrtf(red[0] * (1.0f / C_) + 1e-5f);

    const _Float16* srr = sr + (size_t)row * C_;
    _Float16* zr = z + (size_t)row * C_;
    for (int c = threadIdx.x; c < C_; c += 256) {
        float nv = (yr[c] - mu) * rstd * lnw[c] + lnb[c];
        zr[c] = (_Float16)((float)srr[c] * nv);
    }
}

// ---------------- output WMMA GEMM: out = z @ Wo^T (f32) ----------------
// 16x64 tile per wave: each A fragment feeds 4 WMMAs.
__global__ __launch_bounds__(32)
void gemm_out_kernel(const _Float16* __restrict__ z,
                     const _Float16* __restrict__ Wo,
                     float* __restrict__ out)
{
    const int lane    = threadIdx.x;
    const int rowBase = blockIdx.x * 16;
    const int dBase   = blockIdx.y * (16 * NT_OUT);
    const int m       = lane & 15;
    const int khalf   = (lane >> 4) << 4;

    v8f acc[NT_OUT];
    #pragma unroll
    for (int n = 0; n < NT_OUT; ++n) acc[n] = (v8f)0.0f;

    const _Float16* arow = z + (size_t)(rowBase + m) * C_;
    for (int c0 = 0; c0 < C_; c0 += 32) {
        v16h a = *(const v16h*)(arow + c0 + khalf);
        #pragma unroll
        for (int n = 0; n < NT_OUT; ++n) {
            const int d = dBase + n * 16 + (lane & 15);
            const size_t wo = (size_t)d * C_ + c0 + khalf;
            v16h bb = *(const v16h*)(Wo + wo);
            if (c0 + 32 < C_) __builtin_prefetch(Wo + wo + 32, 0, 3);
            acc[n] = __builtin_amdgcn_wmma_f32_16x16x32_f16(false, a, false, bb,
                        (short)0, acc[n], false, false);
        }
    }
    const int mrow = rowBase + ((lane >> 4) << 3);
    const int ncol = lane & 15;
    #pragma unroll
    for (int n = 0; n < NT_OUT; ++n)
        #pragma unroll
        for (int i = 0; i < 8; ++i)
            out[(size_t)(mrow + i) * C_ + dBase + n * 16 + ncol] = acc[n][i];
}

extern "C" void kernel_launch(void* const* d_in, const int* in_sizes, int n_in,
                              void* d_out, int out_size, void* d_ws, size_t ws_size,
                              hipStream_t stream) {
    const float* x    = (const float*)d_in[0];
    const float* Wk   = (const float*)d_in[1];
    const float* Wv   = (const float*)d_in[2];
    const float* Wr   = (const float*)d_in[3];
    const float* Wo   = (const float*)d_in[4];
    const float* lnw  = (const float*)d_in[5];
    const float* lnb  = (const float*)d_in[6];
    const float* sd   = (const float*)d_in[7];
    const float* sf   = (const float*)d_in[8];
    const float* mixk = (const float*)d_in[9];
    const float* mixv = (const float*)d_in[10];
    const float* mixr = (const float*)d_in[11];
    float* out = (float*)d_out;

    // workspace layout
    char* ws = (char*)d_ws;
    const size_t WB   = (size_t)C_ * C_ * sizeof(_Float16);   // 2 MB per weight
    const size_t BTC  = (size_t)ROWS * C_;
    _Float16* Wk16 = (_Float16*)(ws);
    _Float16* Wv16 = (_Float16*)(ws + WB);
    _Float16* Wr16 = (_Float16*)(ws + 2 * WB);
    _Float16* Wo16 = (_Float16*)(ws + 3 * WB);
    float*    kbuf = (float*)   (ws + 4 * WB);                   // 64 MB (y in-place later)
    float*    vbuf = (float*)   (ws + 4 * WB + BTC * 4);         // 64 MB (z f16 reuses bytes)
    _Float16* srbuf= (_Float16*)(ws + 4 * WB + 2 * BTC * 4);     // 32 MB
    _Float16* zbuf = (_Float16*)vbuf;                            // reuse v buffer bytes

    const int wn = C_ * C_;
    wcvt_kernel<<<wn / 256, 256, 0, stream>>>(Wk, Wk16, wn);
    wcvt_kernel<<<wn / 256, 256, 0, stream>>>(Wv, Wv16, wn);
    wcvt_kernel<<<wn / 256, 256, 0, stream>>>(Wr, Wr16, wn);
    wcvt_kernel<<<wn / 256, 256, 0, stream>>>(Wo, Wo16, wn);

    dim3 ggridKVR(ROWS / 16, C_ / (16 * NT_KVR));
    gemm_kvr_kernel<<<ggridKVR, 32, 0, stream>>>(x, Wk16, Wv16, Wr16,
                                                 mixk, mixv, mixr,
                                                 kbuf, vbuf, srbuf);

    wkv_scan_kernel<<<(B_ * C_) / 256, 256, 0, stream>>>(kbuf, vbuf, sd, sf, kbuf);

    ln_gate_kernel<<<ROWS, 256, 0, stream>>>(kbuf, srbuf, lnw, lnb, zbuf);

    dim3 ggridOUT(ROWS / 16, C_ / (16 * NT_OUT));
    gemm_out_kernel<<<ggridOUT, 32, 0, stream>>>(zbuf, Wo16, out);

    (void)in_sizes; (void)n_in; (void)out_size; (void)ws_size;
}
